// STIXBert_70068096467711
// MI455X (gfx1250) — compile-verified, matrix operand order checked by reading, not run on hardware
//
#include <hip/hip_runtime.h>
#include <math.h>

#define LDSW 136   // LDS row stride in bf16 halves (16B-aligned fragments, bank-skewed)
#define FLAG_GELU 1
#define FLAG_GATE 2

typedef __attribute__((ext_vector_type(16))) __bf16   v16bf;
typedef __attribute__((ext_vector_type(8)))  float    v8f;
typedef __attribute__((ext_vector_type(4)))  unsigned u32x4;
typedef __attribute__((ext_vector_type(4)))  float    f32x4;
// TDM descriptor groups (6-arg builtin: clang-23 / therock-10.0 variant)
typedef __attribute__((ext_vector_type(4))) unsigned int tdm_g0_t;
typedef __attribute__((ext_vector_type(8))) int          tdm_v8i_t;
typedef __attribute__((ext_vector_type(4))) int          tdm_v4i_t;

union FragB { u32x4 u[2]; v16bf v; };

__device__ __forceinline__ unsigned short f2bf(float f) {
  union { float f; unsigned u; } x; x.f = f;
  unsigned r = x.u + 0x7FFFu + ((x.u >> 16) & 1u);   // round-to-nearest-even
  return (unsigned short)(r >> 16);
}
__device__ __forceinline__ float gelu_t(float x) {
  return 0.5f * x * (1.0f + tanhf(0.7978845608028654f * (x + 0.044715f * x * x * x)));
}
// monotonic uint encoding of float so unsigned atomicMax == float max
__device__ __forceinline__ unsigned encf(float f) {
  unsigned u = __float_as_uint(f);
  return (u & 0x80000000u) ? ~u : (u | 0x80000000u);
}
__device__ __forceinline__ float decf(unsigned u) {
  return __uint_as_float((u & 0x80000000u) ? (u & 0x7FFFFFFFu) : ~u);
}

// Y[nrows,128] = op(X[nrows,128]) @ W + b.  Weight comes pre-converted in global
// as transposed bf16 Wt[n][k] (128x128); the Tensor Data Mover DMAs it into LDS
// with pad_interval=64dw / pad_amount=4dw so the LDS stride is LDSW=136 halves.
__global__ __launch_bounds__(256)
void hgt_gemm128(const float* __restrict__ X, const unsigned short* __restrict__ Wt,
                 const float* __restrict__ bias, float* __restrict__ Y,
                 int nrows, int flags, const float* __restrict__ resid,
                 const float* __restrict__ gate)
{
  __shared__ unsigned short sX [64  * LDSW];
  __shared__ unsigned short sWt[128 * LDSW];
  const int tid  = threadIdx.x;
  const int row0 = blockIdx.x * 64;

  // --- wave 0: issue tensor DMA for the weight tile (TENSORcnt tracked) ---
  if (tid < 32) {
    unsigned lds_addr = (unsigned)(size_t)(&sWt[0]);       // LDS aperture: low 32 bits
    unsigned long long ga = (unsigned long long)(size_t)Wt;
    tdm_g0_t g0;
    g0[0] = 1u;                                            // count=1 (valid user D#)
    g0[1] = lds_addr;                                      // lds_addr [63:32]
    g0[2] = (unsigned)(ga & 0xFFFFFFFFu);                  // global_addr [95:64]
    g0[3] = (unsigned)((ga >> 32) & 0x1FFFFFFu)            // global_addr [120:96]
          | 0x80000000u;                                   // type=2 ("image") [127:126]
    tdm_v8i_t g1;
    g1[0] = (1 << 16)      // data_size = 2 bytes (bf16)
          | (1 << 20)      // pad_enable
          | (5 << 22)      // pad_interval: 64 DWORDs (one 128-half row)
          | (3 << 25);     // pad_amount: 4 DWORDs (8 halves -> stride 136)
    g1[1] = (128 << 16);   // tensor_dim0[15:0] = 128
    g1[2] = (128 << 16);   // tensor_dim0 hi=0 | tensor_dim1[15:0] = 128
    g1[3] = (128 << 16);   // tensor_dim1 hi=0 | tile_dim0 = 128
    g1[4] = 128;           // tile_dim1 = 128 | tile_dim2 = 0
    g1[5] = 128;           // tensor_dim0_stride[31:0] = 128 elements
    g1[6] = 0;
    g1[7] = 0;
    tdm_v4i_t gz4 = {0, 0, 0, 0};
    tdm_v8i_t gz8 = {0, 0, 0, 0, 0, 0, 0, 0};
    __builtin_amdgcn_tensor_load_to_lds(g0, g1, gz4, gz4, gz8, 0);
  }

  // --- all threads: stage X tile 64x128 as bf16 (optional GELU) ---
  for (int i = tid; i < 64 * 32; i += 256) {
    int r = i >> 5;
    int c = (i & 31) * 4;
    int grow = row0 + r;
    f32x4 v = {0.f, 0.f, 0.f, 0.f};
    if (grow < nrows) v = *(const f32x4*)(X + (size_t)grow * 128 + c);
    if (grow + 64 < nrows)
      __builtin_prefetch(X + (size_t)(grow + 64) * 128 + c, 0, 1);  // global_prefetch_b8
    if (flags & FLAG_GELU) { v.x=gelu_t(v.x); v.y=gelu_t(v.y); v.z=gelu_t(v.z); v.w=gelu_t(v.w); }
    unsigned short* dst = &sX[r * LDSW + c];
    dst[0]=f2bf(v.x); dst[1]=f2bf(v.y); dst[2]=f2bf(v.z); dst[3]=f2bf(v.w);
  }
  if (tid < 32) __builtin_amdgcn_s_wait_tensorcnt(0);
  __syncthreads();

  const int lane  = tid & 31;
  const int wave  = tid >> 5;
  const int m15   = lane & 15;
  const int khalf = lane >> 4;
  const int rt      = wave >> 1;        // row tile 0..3 (16 rows each)
  const int colbase = (wave & 1) * 64;  // 4 col tiles of 16

  v8f acc0 = {}, acc1 = {}, acc2 = {}, acc3 = {};
  const unsigned short* aRow = &sX [(rt * 16 + m15) * LDSW];
  const unsigned short* bR0  = &sWt[(colbase +  0 + m15) * LDSW];
  const unsigned short* bR1  = &sWt[(colbase + 16 + m15) * LDSW];
  const unsigned short* bR2  = &sWt[(colbase + 32 + m15) * LDSW];
  const unsigned short* bR3  = &sWt[(colbase + 48 + m15) * LDSW];

#pragma unroll
  for (int ks = 0; ks < 4; ++ks) {
    const int off = ks * 32 + khalf * 8;   // per-ISA 16-bit A 16x32 fragment layout
    FragB af;
    { const u32x4* p = (const u32x4*)(aRow + off); af.u[0] = p[0]; af.u[1] = p[2]; }
    FragB b0, b1, b2, b3;
    { const u32x4* p = (const u32x4*)(bR0 + off); b0.u[0]=p[0]; b0.u[1]=p[2]; }
    { const u32x4* p = (const u32x4*)(bR1 + off); b1.u[0]=p[0]; b1.u[1]=p[2]; }
    { const u32x4* p = (const u32x4*)(bR2 + off); b2.u[0]=p[0]; b2.u[1]=p[2]; }
    { const u32x4* p = (const u32x4*)(bR3 + off); b3.u[0]=p[0]; b3.u[1]=p[2]; }
    acc0 = __builtin_amdgcn_wmma_f32_16x16x32_bf16(false, af.v, false, b0.v, (short)0, acc0, false, false);
    acc1 = __builtin_amdgcn_wmma_f32_16x16x32_bf16(false, af.v, false, b1.v, (short)0, acc1, false, false);
    acc2 = __builtin_amdgcn_wmma_f32_16x16x32_bf16(false, af.v, false, b2.v, (short)0, acc2, false, false);
    acc3 = __builtin_amdgcn_wmma_f32_16x16x32_bf16(false, af.v, false, b3.v, (short)0, acc3, false, false);
  }

  float gg = 1.0f, gi = 0.0f;
  if (flags & FLAG_GATE) { gg = 1.0f / (1.0f + __expf(-gate[0])); gi = 1.0f - gg; }
  v8f accs[4] = {acc0, acc1, acc2, acc3};
#pragma unroll
  for (int ct = 0; ct < 4; ++ct) {
    int n = colbase + ct * 16 + m15;
    float bv = bias ? bias[n] : 0.0f;
#pragma unroll
    for (int r = 0; r < 8; ++r) {
      int grow = row0 + rt * 16 + khalf * 8 + r;   // D: lanes16-31 hold M=8+r
      if (grow < nrows) {
        float val = accs[ct][r] + bv;
        if (flags & FLAG_GATE) val = gg * val + gi * resid[(size_t)grow * 128 + n];
        Y[(size_t)grow * 128 + n] = val;
      }
    }
  }
}

// f32 W[k][n] -> transposed bf16 Wt[n][k] in global (TDM source)
__global__ void prep_w(const float* __restrict__ Wg, unsigned short* __restrict__ Wt) {
  int i = blockIdx.x * 256 + threadIdx.x;
  if (i >= 128 * 128) return;
  int k = i >> 7, n = i & 127;
  Wt[n * 128 + k] = f2bf(Wg[i]);
}

// per-head [4,32,32] rel matrix -> block-diagonal transposed bf16 Wt[n][k]
__global__ void prep_bd(const float* __restrict__ rel, unsigned short* __restrict__ Wt) {
  int i = blockIdx.x * 256 + threadIdx.x;
  if (i >= 128 * 128) return;
  int k = i >> 7, n = i & 127;
  int hk = k >> 5, hn = n >> 5;
  Wt[n * 128 + k] = (hk == hn) ? f2bf(rel[hk * 1024 + (k & 31) * 32 + (n & 31)])
                               : (unsigned short)0;
}

__global__ void zero_f(float* __restrict__ p, size_t n) {
  size_t i = (size_t)blockIdx.x * blockDim.x + threadIdx.x;
  size_t st = (size_t)gridDim.x * blockDim.x;
  for (; i < n; i += st) p[i] = 0.0f;
}

// pass A: s = (q[di] . kr[si]) per head * p_rel / sqrt(32); atomicMax per (node,head)
__global__ void edge_score(const float* __restrict__ q, const float* __restrict__ kr,
                           const int* __restrict__ ei, int E,
                           const float* __restrict__ prel,
                           float* __restrict__ sbuf, unsigned* __restrict__ mmax)
{
  int idx = blockIdx.x * 256 + threadIdx.x;
  if (idx >= E * 4) return;
  int e = idx >> 2, hh = idx & 3;
  int si = ei[e], di = ei[E + e];
  const f32x4* qp = (const f32x4*)(q  + (size_t)di * 128 + hh * 32);
  const f32x4* kp = (const f32x4*)(kr + (size_t)si * 128 + hh * 32);
  float dot = 0.f;
#pragma unroll
  for (int i = 0; i < 8; ++i) {
    f32x4 a = qp[i], b = kp[i];
    dot += a.x * b.x + a.y * b.y + a.z * b.z + a.w * b.w;
  }
  float s = dot * prel[hh] * 0.17677669529663687f;  // 1/sqrt(32)
  sbuf[idx] = s;
  atomicMax(&mmax[(size_t)di * 4 + hh], encf(s));
}

// pass B: ex = exp(s - m); den += ex
__global__ void edge_exp(const int* __restrict__ ei, int E,
                         float* __restrict__ sbuf, const unsigned* __restrict__ mmax,
                         float* __restrict__ den)
{
  int idx = blockIdx.x * 256 + threadIdx.x;
  if (idx >= E * 4) return;
  int e = idx >> 2, hh = idx & 3;
  int di = ei[E + e];
  float m = decf(mmax[(size_t)di * 4 + hh]);
  float ex = __expf(sbuf[idx] - m);
  sbuf[idx] = ex;
  atomicAdd(&den[(size_t)di * 4 + hh], ex);
}

// pass C: agg[di] += (ex/den) * vr[si]
__global__ void edge_agg(const int* __restrict__ ei, int E,
                         const float* __restrict__ sbuf, const float* __restrict__ den,
                         const float* __restrict__ vr, float* __restrict__ agg)
{
  int idx = blockIdx.x * 256 + threadIdx.x;
  if (idx >= E * 4) return;
  int e = idx >> 2, hh = idx & 3;
  int si = ei[e], di = ei[E + e];
  float alpha = sbuf[idx] / den[(size_t)di * 4 + hh];
  const f32x4* vp = (const f32x4*)(vr + (size_t)si * 128 + hh * 32);
  float* ag = agg + (size_t)di * 128 + hh * 32;
#pragma unroll
  for (int i = 0; i < 8; ++i) {
    f32x4 vv = vp[i];
    atomicAdd(&ag[i * 4 + 0], alpha * vv.x);
    atomicAdd(&ag[i * 4 + 1], alpha * vv.y);
    atomicAdd(&ag[i * 4 + 2], alpha * vv.z);
    atomicAdd(&ag[i * 4 + 3], alpha * vv.w);
  }
}

// h = LN(hn + h) * g + b   (one wave32 per 128-wide row)
__global__ __launch_bounds__(256)
void resid_ln(const float* __restrict__ hn, float* __restrict__ h,
              const float* __restrict__ g, const float* __restrict__ b, int nrows)
{
  int row  = blockIdx.x * 8 + (threadIdx.x >> 5);
  int lane = threadIdx.x & 31;
  if (row >= nrows) return;
  size_t base = (size_t)row * 128 + lane * 4;
  f32x4 a = *(const f32x4*)(hn + base);
  f32x4 c = *(const f32x4*)(h  + base);
  f32x4 x = a + c;
  float sum = x.x + x.y + x.z + x.w;
  for (int o = 16; o > 0; o >>= 1) sum += __shfl_xor(sum, o, 32);
  float mean = sum * (1.0f / 128.0f);
  f32x4 d = x - mean;
  float vs = d.x * d.x + d.y * d.y + d.z * d.z + d.w * d.w;
  for (int o = 16; o > 0; o >>= 1) vs += __shfl_xor(vs, o, 32);
  float inv = rsqrtf(vs * (1.0f / 128.0f) + 1e-5f);
  f32x4 gg = *(const f32x4*)(g + lane * 4);
  f32x4 bb = *(const f32x4*)(b + lane * 4);
  f32x4 out = d * inv * gg + bb;
  *(f32x4*)(h + base) = out;
}

extern "C" void kernel_launch(void* const* d_in, const int* in_sizes, int n_in,
                              void* d_out, int out_size, void* d_ws, size_t ws_size,
                              hipStream_t stream) {
  (void)in_sizes; (void)n_in; (void)out_size; (void)ws_size;
  const int N = 100000, E = 500000;
  const size_t NF = (size_t)N * 128;

  auto P = [&](int i) { return (const float*)d_in[i]; };

  // param leaves (jax pytree: dict keys sorted). Alpha order:
  // 0=attack_pattern 1=indicator 2=malware 3=threat_actor
  // in_proj: [0..7] (b,w per type); layer l base L0 = 8 + 56*l:
  //   a: L0+2t(b),+1(w) | k: L0+8+2t,+1 | ln: L0+16+2t(b),+1(g)
  //   q: L0+24+2t,+1    | rel: L0+32+3r (a_rel,m_rel,p_rel)
  //   skip: L0+44+t     | v: L0+48+2t,+1
  // out_proj: 120+2t(b),+1(w); x: 128..131; ei: 132..135
  const float* x[4];
  x[0] = P(131); x[1] = P(129); x[2] = P(128); x[3] = P(130);
  const int* ei[4];
  for (int r = 0; r < 4; ++r) ei[r] = (const int*)d_in[132 + r];
  const int RSRC[4] = {1, 3, 2, 3};   // src alpha-type per relation
  const int RDST[4] = {2, 2, 0, 0};   // dst alpha-type per relation

  // workspace layout (floats)
  float* w = (float*)d_ws;
  float *h[4], *hn[4], *kb[4], *qb[4], *vb[4], *aggb[4];
  for (int t = 0; t < 4; ++t) {
    h[t]    = w + (size_t)t * NF;
    hn[t]   = w + (size_t)(4 + t) * NF;
    kb[t]   = w + (size_t)(8 + t) * NF;
    qb[t]   = w + (size_t)(12 + t) * NF;
    vb[t]   = w + (size_t)(16 + t) * NF;
    aggb[t] = w + (size_t)(22 + t) * NF;
  }
  float* krb = w + (size_t)20 * NF;
  float* vrb = w + (size_t)21 * NF;
  float* sbuf[4];
  for (int r = 0; r < 4; ++r) sbuf[r] = w + (size_t)26 * NF + (size_t)r * E * 4;
  float* mmbase  = w + (size_t)26 * NF + (size_t)4 * E * 4;
  float* denbase = mmbase + (size_t)16 * N;
  unsigned short* wt = (unsigned short*)(denbase + (size_t)16 * N);  // 128x128 bf16
  unsigned* mm[4]; float* den[4];
  for (int t = 0; t < 4; ++t) {
    mm[t]  = (unsigned*)(mmbase + (size_t)t * 4 * N);
    den[t] = denbase + (size_t)t * 4 * N;
  }

  const dim3 gB(256);
  const dim3 gGemm((N + 63) / 64);
  const dim3 gEdge((E * 4 + 255) / 256);
  auto prepw  = [&](const float* Wg)  { prep_w <<<64, gB, 0, stream>>>(Wg,  wt); };
  auto prepbd = [&](const float* rel) { prep_bd<<<64, gB, 0, stream>>>(rel, wt); };
  auto gemm = [&](const float* X, const float* bias, float* Y,
                  int flags, const float* resid, const float* gate) {
    hgt_gemm128<<<gGemm, gB, 0, stream>>>(X, wt, bias, Y, N, flags, resid, gate);
  };

  // input projection
  for (int t = 0; t < 4; ++t) {
    prepw(P(2 * t + 1));
    gemm(x[t], P(2 * t), h[t], 0, nullptr, nullptr);
  }

  for (int l = 0; l < 2; ++l) {
    const int L0 = 8 + 56 * l;
    // K/Q/V projections
    for (int t = 0; t < 4; ++t) {
      prepw(P(L0 + 8  + 2 * t + 1)); gemm(h[t], P(L0 + 8  + 2 * t), kb[t], 0, nullptr, nullptr);
      prepw(P(L0 + 24 + 2 * t + 1)); gemm(h[t], P(L0 + 24 + 2 * t), qb[t], 0, nullptr, nullptr);
      prepw(P(L0 + 48 + 2 * t + 1)); gemm(h[t], P(L0 + 48 + 2 * t), vb[t], 0, nullptr, nullptr);
    }
    // clear agg / segment-max / denominators
    zero_f<<<2048, gB, 0, stream>>>(aggb[0], (size_t)4 * NF);
    zero_f<<<64,   gB, 0, stream>>>(mmbase, (size_t)32 * N);

    // pass A: kr = k[src] @ blockdiag(a_rel); scores + segment max
    for (int r = 0; r < 4; ++r) {
      prepbd(P(L0 + 32 + 3 * r));
      gemm(kb[RSRC[r]], nullptr, krb, 0, nullptr, nullptr);
      edge_score<<<gEdge, gB, 0, stream>>>(qb[RDST[r]], krb, ei[r], E,
                                           P(L0 + 32 + 3 * r + 2), sbuf[r], mm[RDST[r]]);
    }
    // pass B: exp + denominators
    for (int r = 0; r < 4; ++r)
      edge_exp<<<gEdge, gB, 0, stream>>>(ei[r], E, sbuf[r], mm[RDST[r]], den[RDST[r]]);
    // pass C: vr = v[src] @ blockdiag(m_rel); weighted scatter-add
    for (int r = 0; r < 4; ++r) {
      prepbd(P(L0 + 32 + 3 * r + 1));
      gemm(vb[RSRC[r]], nullptr, vrb, 0, nullptr, nullptr);
      edge_agg<<<gEdge, gB, 0, stream>>>(ei[r], E, sbuf[r], den[RDST[r]], vrb, aggb[RDST[r]]);
    }
    // a-proj with fused GELU input and sigmoid skip-gate blend
    for (int t = 0; t < 4; ++t) {
      prepw(P(L0 + 2 * t + 1));
      gemm(aggb[t], P(L0 + 2 * t), hn[t], FLAG_GELU | FLAG_GATE, h[t], P(L0 + 44 + t));
    }
    // residual + LayerNorm
    for (int t = 0; t < 4; ++t)
      resid_ln<<<(N + 7) / 8, gB, 0, stream>>>(hn[t], h[t],
                                               P(L0 + 16 + 2 * t + 1), P(L0 + 16 + 2 * t), N);
  }

  // output projection, reference NODE_TYPES order: malware, indicator, threat_actor, attack_pattern
  const int OUT_ALPHA[4] = {2, 1, 3, 0};
  float* out = (float*)d_out;
  for (int pos = 0; pos < 4; ++pos) {
    int t = OUT_ALPHA[pos];
    prepw(P(120 + 2 * t + 1));
    gemm(h[t], P(120 + 2 * t), out + (size_t)pos * NF, 0, nullptr, nullptr);
  }
}
